// MSACrossAttention_5360119185745
// MI455X (gfx1250) — matrix operand order repaired and checked by reading, not compiled, not run on hardware
//
#include <hip/hip_runtime.h>
#include <hip/hip_bf16.h>

// ---------------------------------------------------------------------------
// MSACrossAttention, M=1 degenerate form (Q is dead code in the reference):
//   Vr[r, a]  = dot(W_kv[(r&15)*128+64+a, :], x[r>>4, :]) + b_kv[...]   (r = s*16+c)
//   out[r, d] = rmsnorm_d( Vr[r,:]·W_o[d,:] + b_o[d] + x[r&4095, d] ) * norm_w[d]
// out is (16, 4096, 1024) f32 = 256 MB -> bandwidth bound (~12.4 us @ 23.3 TB/s).
// Matmuls in bf16 WMMA (v_wmma_f32_16x16x32_bf16), f32 accumulate.
// Stage-1 GEMM uses 32x32 register blocking per wave: 4 WMMA per 8 b128 loads,
// 4 independent accumulator chains.
// ---------------------------------------------------------------------------

typedef __attribute__((ext_vector_type(16))) __bf16 v16bf;
typedef __attribute__((ext_vector_type(8)))  __bf16 v8bf;
typedef __attribute__((ext_vector_type(8)))  float  v8f;

__device__ __forceinline__ unsigned short f2bf(float f) {
    unsigned u = __float_as_uint(f);
    u += 0x7FFFu + ((u >> 16) & 1u);        // round-to-nearest-even
    return (unsigned short)(u >> 16);
}
__device__ __forceinline__ unsigned pack2(float a, float b) {
    return (unsigned)f2bf(a) | ((unsigned)f2bf(b) << 16);
}

// A 16x32 bf16 fragment: lane(l15)=row M, hi selects K 0..15 / 16..31
__device__ __forceinline__ v16bf load_afrag(const unsigned short* row, int k0, int hi) {
    return *reinterpret_cast<const v16bf*>(row + k0 + hi * 16);
}
// B 32x16 bf16 fragment: lane(l15)=col N, hi selects K 0..15 / 16..31
__device__ __forceinline__ v16bf load_bfrag(const unsigned short* row, int k0, int hi) {
    return *reinterpret_cast<const v16bf*>(row + k0 + hi * 16);
}
__device__ __forceinline__ v8f wmma_bf16(v16bf a, v16bf b, v8f c) {
    return __builtin_amdgcn_wmma_f32_16x16x32_bf16(false, a, false, b,
                                                   (short)0, c, false, false);
}

// ---- Stage 0a: x (4096x1024 f32) -> bf16 ----------------------------------
__global__ void msa_cvt_x(const float* __restrict__ in, unsigned short* __restrict__ outp) {
    unsigned i = (blockIdx.x * blockDim.x + threadIdx.x) * 4u;
    float4 v = *reinterpret_cast<const float4*>(in + i);
    uint2 p;
    p.x = pack2(v.x, v.y);
    p.y = pack2(v.z, v.w);
    *reinterpret_cast<uint2*>(outp + i) = p;
}

// ---- Stage 0b: V-half rows of W_kv, remapped: wvb[j,d] = W_kv[(j>>6)*128+64+(j&63), d]
__global__ void msa_cvt_wv(const float* __restrict__ wkv, unsigned short* __restrict__ outp) {
    unsigned i = (blockIdx.x * blockDim.x + threadIdx.x) * 4u;   // dest element idx
    unsigned j = i >> 10;
    unsigned d = i & 1023u;
    unsigned src = ((j >> 6) * 128u + 64u + (j & 63u)) * 1024u + d;
    float4 v = *reinterpret_cast<const float4*>(wkv + src);
    uint2 p;
    p.x = pack2(v.x, v.y);
    p.y = pack2(v.z, v.w);
    *reinterpret_cast<uint2*>(outp + i) = p;
}

// ---- Stage 0c: W_o (1024x64 f32) -> bf16 (same layout) --------------------
__global__ void msa_cvt_wo(const float* __restrict__ in, unsigned short* __restrict__ outp) {
    unsigned i = (blockIdx.x * blockDim.x + threadIdx.x) * 4u;
    float4 v = *reinterpret_cast<const float4*>(in + i);
    uint2 p;
    p.x = pack2(v.x, v.y);
    p.y = pack2(v.z, v.w);
    *reinterpret_cast<uint2*>(outp + i) = p;
}

// ---- Stage 1: V GEMM: V(4096x1024) = Xb(4096x1024) @ Wvb^T, K=1024 --------
// One wave = 32x32 output (2x2 WMMA tiles). 128 x 32 wave-tiles = 4096 waves.
// The 8 waves of a block share the same 32-row group (A rows hit WGP$).
__global__ void msa_gemm_v(const unsigned short* __restrict__ xb,
                           const unsigned short* __restrict__ wvb,
                           const float* __restrict__ bkv,
                           unsigned short* __restrict__ vr) {
    const int wave = blockIdx.x * (blockDim.x >> 5) + (threadIdx.x >> 5);
    const int lane = threadIdx.x & 31;
    const int mt   = wave >> 5;        // 0..127  (32-row group)
    const int nt   = wave & 31;        // 0..31   (32-col group)
    const int hi   = lane >> 4;
    const int l15  = lane & 15;

    const unsigned short* arow0 = xb  + (mt * 32 + l15) * 1024;
    const unsigned short* arow1 = arow0 + 16 * 1024;
    const unsigned short* brow0 = wvb + (nt * 32 + l15) * 1024;
    const unsigned short* brow1 = brow0 + 16 * 1024;

    v8f acc00 = {}, acc01 = {}, acc10 = {}, acc11 = {};

    for (int kb = 0; kb < 1024; kb += 256) {
        // speculative prefetch of the next 512B per stream (drops past the end)
        __builtin_prefetch(arow0 + kb + 256, 0, 3);   // global_prefetch_b8
        __builtin_prefetch(arow1 + kb + 256, 0, 3);
        __builtin_prefetch(brow0 + kb + 256, 0, 3);
        __builtin_prefetch(brow1 + kb + 256, 0, 3);
#pragma unroll
        for (int ko = 0; ko < 256; ko += 32) {
            const int k0 = kb + ko;
            v16bf A0 = load_afrag(arow0, k0, hi);
            v16bf A1 = load_afrag(arow1, k0, hi);
            v16bf B0 = load_bfrag(brow0, k0, hi);
            v16bf B1 = load_bfrag(brow1, k0, hi);
            acc00 = wmma_bf16(A0, B0, acc00);
            acc01 = wmma_bf16(A0, B1, acc01);
            acc10 = wmma_bf16(A1, B0, acc10);
            acc11 = wmma_bf16(A1, B1, acc11);
        }
    }

    // Epilogue: + b_kv, convert to bf16, store into Vr (65536x64).
#pragma unroll
    for (int i = 0; i < 2; ++i) {
#pragma unroll
        for (int j = 0; j < 2; ++j) {
            const v8f acc = (i == 0) ? (j == 0 ? acc00 : acc01)
                                     : (j == 0 ? acc10 : acc11);
            const int jcol = nt * 32 + j * 16 + l15;       // V column 0..1023
            const float bk = bkv[(jcol >> 6) * 128 + 64 + (jcol & 63)];
            const int vcol = jcol & 63;
            const int coff = jcol >> 6;
#pragma unroll
            for (int jj = 0; jj < 8; ++jj) {
                const int s = mt * 32 + i * 16 + hi * 8 + jj;   // D row M = hi*8+jj
                vr[(s * 16 + coff) * 64 + vcol] = f2bf(acc[jj] + bk);
            }
        }
    }
}

// ---- Stage 2: fused Out = rmsnorm(Vr @ Wo^T + b_o + x_resid) * norm_w -----
// Block = 16 output rows x 1024 cols (8 waves x 8 tiles, K=64 -> 2 WMMAs/tile).
__global__ void msa_out(const unsigned short* __restrict__ vr,
                        const unsigned short* __restrict__ wob,
                        const float* __restrict__ x,
                        const float* __restrict__ bo,
                        const float* __restrict__ nw,
                        float* __restrict__ out) {
    __shared__ float ssq[16];
    __shared__ float rinv[16];
    const int tid   = threadIdx.x;
    const int wv    = tid >> 5;
    const int lane  = tid & 31;
    const int hi    = lane >> 4;
    const int l15   = lane & 15;
    const int rbase = blockIdx.x * 16;          // output row tile
    const int sbase = rbase & 4095;             // residual row base

    if (tid < 16) ssq[tid] = 0.0f;
    __syncthreads();

    const unsigned short* arow = vr + (rbase + l15) * 64;   // shared by all 8 waves

    float yreg[8][8];
    float part[8] = {0.f, 0.f, 0.f, 0.f, 0.f, 0.f, 0.f, 0.f};

#pragma unroll
    for (int t = 0; t < 8; ++t) {
        const int dcol = (wv * 8 + t) * 16 + l15;
        const unsigned short* brow = wob + dcol * 64;
        v8f acc = {};
#pragma unroll
        for (int k0 = 0; k0 < 64; k0 += 32) {
            v16bf A = load_afrag(arow, k0, hi);
            v16bf B = load_bfrag(brow, k0, hi);
            acc = wmma_bf16(A, B, acc);
        }
        const float bod = bo[dcol];
#pragma unroll
        for (int j = 0; j < 8; ++j) {
            const int row = hi * 8 + j;
            const float y = acc[j] + bod + x[(sbase + row) * 1024 + dcol];
            yreg[t][j] = y;
            part[j] += y * y;
        }
    }

#pragma unroll
    for (int j = 0; j < 8; ++j)
        atomicAdd(&ssq[hi * 8 + j], part[j]);    // ds_add_f32
    __syncthreads();

    if (tid < 16) rinv[tid] = rsqrtf(ssq[tid] * (1.0f / 1024.0f) + 1e-6f);
    __syncthreads();

#pragma unroll
    for (int t = 0; t < 8; ++t) {
        const int dcol = (wv * 8 + t) * 16 + l15;
        const float w = nw[dcol];
#pragma unroll
        for (int j = 0; j < 8; ++j) {
            const int row = hi * 8 + j;
            out[(rbase + row) * 1024 + dcol] = yreg[t][j] * rinv[row] * w;
        }
    }
}

// ---------------------------------------------------------------------------
extern "C" void kernel_launch(void* const* d_in, const int* in_sizes, int n_in,
                              void* d_out, int out_size, void* d_ws, size_t ws_size,
                              hipStream_t stream) {
    (void)in_sizes; (void)n_in; (void)out_size; (void)ws_size;
    const float* x   = (const float*)d_in[0];
    // d_in[1] (W_q), d_in[2] (b_q): dead in the reference graph
    const float* wkv = (const float*)d_in[3];
    const float* bkv = (const float*)d_in[4];
    const float* wo  = (const float*)d_in[5];
    const float* bo  = (const float*)d_in[6];
    const float* nw  = (const float*)d_in[7];
    float* out = (float*)d_out;

    char* ws = (char*)d_ws;
    unsigned short* xb  = (unsigned short*)(ws);                          // 8 MB
    unsigned short* wvb = (unsigned short*)(ws + (size_t)(8u  << 20));    // 2 MB
    unsigned short* wob = (unsigned short*)(ws + (size_t)(10u << 20));    // 128 KB
    unsigned short* vr  = (unsigned short*)(ws + (size_t)(11u << 20));    // 8 MB

    msa_cvt_x <<<4096, 256, 0, stream>>>(x,   xb);    // 4,194,304 elems
    msa_cvt_wv<<<1024, 256, 0, stream>>>(wkv, wvb);   // 1,048,576 elems
    msa_cvt_wo<<<  64, 256, 0, stream>>>(wo,  wob);   //    65,536 elems
    msa_gemm_v<<< 512, 256, 0, stream>>>(xb, wvb, bkv, vr);   // 4096 waves, 32x32/wave
    msa_out   <<<4096, 256, 0, stream>>>(vr, wob, x, bo, nw, out);
}
